// UpdateONNX_successul_on_AMBA_20134806683719
// MI455X (gfx1250) — matrix operand order repaired and checked by reading, not compiled, not run on hardware
//
#include <hip/hip_runtime.h>

// ---------------------------------------------------------------------------
// Fused residual-MLP update for MI455X (gfx1250, wave32, WMMA f16->f32).
//
//   x   = net + inp + Lin3(relu(Lin2(relu(Lin1(corr)))))     (882->384->384->384)
//   x2  = x + Lin(relu(Lin(x)))  [c1]  + Lin(relu(Lin(x)))   [c2]
//   out = (x2, Head_d(relu(x2)), sigmoid(Head_w(relu(x2))))
//
// All GEMMs run on V_WMMA_F32_16X16X32_F16.  Weights are pre-packed (per call)
// into f16 B-fragment lane order in d_ws.  Activations stay in LDS as f16
// A-fragments; the residual accumulator stays in LDS as f32.
// ---------------------------------------------------------------------------

#define DIMC  384
#define HH    65536
#define CORRK 882
#define KPAD  896      // corr K padded to multiple of 32
#define KT1   28       // K-tiles for layer 1 (896/32)
#define KTD   12       // K-tiles for 384-wide layers
#define NTT   24       // N-tiles (384/16)
#define HB    64       // rows per workgroup

typedef __attribute__((ext_vector_type(16))) _Float16 v16h;
typedef __attribute__((ext_vector_type(8)))  float    v8f;
typedef __attribute__((ext_vector_type(4)))  float    v4f;

// ---- LDS layout (bytes). Regions aliased by liveness. Total 256 KB. -------
#define LDS_T2   0        // [0,114688): corr A-frags (stage0/L1); then t2 (48K) + u (48K)
#define LDS_U    49152
#define LDS_T1   114688   // 48K: t1 A-frags; later xA
#define LDS_XA   114688
#define LDS_BSUM 163840   // 96K: f32 residual, index [n*64 + m]
#define LDS_TOTAL 262144

// ---- packed-weight offsets in d_ws (units: halves) ------------------------
#define OFF_W1 0          // cc_w1: 24*28*512 = 344064
#define OFF_W2 344064     // each 384x384: 24*12*512 = 147456
#define OFF_W3 491520
#define OFF_W4 638976
#define OFF_W5 786432
#define OFF_W6 933888
#define OFF_W7 1081344    // end = 1228800 halves = 2.34 MB in d_ws

// ---------------------------------------------------------------------------
// Weight pre-pack: f32 row-major W[Cout][Cin]  ->  f16 B-fragments (32x16 KxN
// tiles).  ISA B layout (16-bit, wave32): lane l holds N = l&15, K-group
// (l>>4)*8; vector element e (VGPR e/2, half e&1) holds K = group + (e&7) + 16*(e>>3).
// Destination is written per-lane-contiguous so the GEMM loads one v16h/lane.
// ---------------------------------------------------------------------------
__global__ void __launch_bounds__(256)
pack_w_kernel(const float* __restrict__ W, _Float16* __restrict__ dst,
              int Kreal, int Ktiles)
{
    const int f    = blockIdx.x * 256 + threadIdx.x;   // grid covers total exactly
    const int tile = f >> 9;                           // nt*Ktiles + kt
    const int w    = f & 511;
    const int lane = w >> 4;
    const int e    = w & 15;
    const int nt   = tile / Ktiles;
    const int kt   = tile - nt * Ktiles;
    const int n    = (nt << 4) + (lane & 15);
    const int k    = (kt << 5) + ((e >> 3) << 4) + ((lane >> 4) << 3) + (e & 7);
    dst[f] = (k < Kreal) ? (_Float16)W[n * Kreal + k] : (_Float16)0.f;
}

// ---------------------------------------------------------------------------
// Scatter one C-layout f32 tile column (8 rows held by this lane) into the
// f16 A-fragment layout of the next layer (ISA 16-bit 16x32 A layout):
//   k -> lane-group g2=(k>>3)&1, vgpr i=((k>>4)<<2)|((k>>1)&3), half h=k&1
// ---------------------------------------------------------------------------
__device__ __forceinline__ void scatter_afrag(char* smem, int dOff, int dKt, int mt,
                                              int n, int g, const float* x, bool relu)
{
    const int kt = n >> 5, kk = n & 31;
    const int g2 = (kk >> 3) & 1;
    const int ii = ((kk >> 4) << 2) | ((kk >> 1) & 3);
    const int h2 = kk & 1;
    _Float16* base = (_Float16*)(smem + dOff);
    const int tb = ((mt * dKt + kt) << 9) + (ii << 1) + h2;
#pragma unroll
    for (int r = 0; r < 8; ++r) {
        float v = x[r];
        if (relu) v = fmaxf(v, 0.f);
        const int m = r + (g << 3);
        base[tb + ((m + (g2 << 4)) << 4)] = (_Float16)v;
    }
}

// MODE 0: bias(+relu) -> scatter f16 A-frags at dOff
// MODE 1: bias + net + inp -> bsum (f32) and scatter f16 xA (no relu)
// MODE 2: bsum += acc + bias
template <int MODE, bool RELU>
__device__ __forceinline__ void epilogue(char* smem, v8f acc, int nt, int mt, int lane,
                                         const float* __restrict__ bias,
                                         int dOff, int dKt,
                                         const float* __restrict__ netp,
                                         const float* __restrict__ inpp, int h0)
{
    const int nlocal = lane & 15;
    const int g      = lane >> 4;          // C layout: lane holds N=nlocal, M=r+8g
    const int n      = (nt << 4) + nlocal;
    const float bv   = bias[n];
    float x[8];
#pragma unroll
    for (int r = 0; r < 8; ++r) x[r] = acc[r] + bv;

    if (MODE == 1) {
        const v4f* np = (const v4f*)(netp + (size_t)n * HH + h0 + (g << 3));
        const v4f* ip = (const v4f*)(inpp + (size_t)n * HH + h0 + (g << 3));
        const v4f n0 = np[0], n1 = np[1], i0 = ip[0], i1 = ip[1];
#pragma unroll
        for (int r = 0; r < 4; ++r) { x[r] += n0[r] + i0[r]; x[r + 4] += n1[r] + i1[r]; }
        float* bs = (float*)(smem + LDS_BSUM) + n * HB + (g << 3);
        v4f s0 = {x[0], x[1], x[2], x[3]};
        v4f s1 = {x[4], x[5], x[6], x[7]};
        *(v4f*)(bs)     = s0;
        *(v4f*)(bs + 4) = s1;
    }
    if (MODE == 2) {
        float* bs = (float*)(smem + LDS_BSUM) + n * HB + (g << 3);
        v4f s0 = *(v4f*)(bs), s1 = *(v4f*)(bs + 4);
#pragma unroll
        for (int r = 0; r < 4; ++r) { s0[r] += x[r]; s1[r] += x[r + 4]; }
        *(v4f*)(bs)     = s0;
        *(v4f*)(bs + 4) = s1;
        return;
    }
    scatter_afrag(smem, dOff, dKt, mt, n, g, x, RELU);   // MODE 0 / 1
}

// One 16(M) x 192(N) x (32*KTILES)(K) GEMM slice per wave, dual WMMA chains.
template <int KTILES, int MODE, bool RELU>
__device__ __forceinline__ void layer_gemm(char* smem, int aOff,
                                           const _Float16* __restrict__ W,
                                           const float* __restrict__ bias,
                                           int mt, int nh, int lane,
                                           int dOff, int dKt,
                                           const float* __restrict__ netp,
                                           const float* __restrict__ inpp, int h0)
{
    for (int nt2 = 0; nt2 < 6; ++nt2) {
        const int ntA = nh * 12 + nt2 * 2;
        const int ntB = ntA + 1;
        v8f acc0 = {};
        v8f acc1 = {};
#pragma unroll
        for (int kt = 0; kt < KTILES; ++kt) {
            const v16h a  = *(const v16h*)(smem + aOff + ((mt * KTILES + kt) << 10) + (lane << 5));
            const v16h b0 = *(const v16h*)(W + ((ntA * KTILES + kt) << 9) + (lane << 4));
            const v16h b1 = *(const v16h*)(W + ((ntB * KTILES + kt) << 9) + (lane << 4));
            acc0 = __builtin_amdgcn_wmma_f32_16x16x32_f16(false, a, false, b0, (short)0, acc0, false, false);
            acc1 = __builtin_amdgcn_wmma_f32_16x16x32_f16(false, a, false, b1, (short)0, acc1, false, false);
        }
        epilogue<MODE, RELU>(smem, acc0, ntA, mt, lane, bias, dOff, dKt, netp, inpp, h0);
        epilogue<MODE, RELU>(smem, acc1, ntB, mt, lane, bias, dOff, dKt, netp, inpp, h0);
    }
}

// ---------------------------------------------------------------------------
__global__ void __launch_bounds__(256)
fused_update_kernel(const float* __restrict__ net, const float* __restrict__ inp,
                    const float* __restrict__ corr, const _Float16* __restrict__ ws,
                    const float* __restrict__ b1v, const float* __restrict__ b2v,
                    const float* __restrict__ b3v, const float* __restrict__ b4v,
                    const float* __restrict__ b5v, const float* __restrict__ b6v,
                    const float* __restrict__ b7v,
                    const float* __restrict__ dW, const float* __restrict__ dB,
                    const float* __restrict__ wW, const float* __restrict__ wB,
                    float* __restrict__ out)
{
    extern __shared__ char smem[];
    const int h0   = blockIdx.x * HB;
    const int tid  = threadIdx.x;
    const int wave = tid >> 5, lane = tid & 31;
    const int mt   = wave & 3;      // M-tile (16 rows) owned by this wave
    const int nh   = wave >> 2;     // N-half (192 cols) owned by this wave

    // ---- Stage 0: corr [882 x H] (channel-major) -> f16 A-frags in LDS ----
    for (int it = tid; it < KPAD * 16; it += 256) {
        const int k  = it >> 4;
        const int h4 = (it & 15) << 2;      // 4 consecutive rows
        v4f v = {0.f, 0.f, 0.f, 0.f};
        if (k < CORRK) v = *(const v4f*)(corr + (size_t)k * HH + h0 + h4);
        const int kt = k >> 5, kk = k & 31;
        const int g2 = (kk >> 3) & 1;
        const int ii = ((kk >> 4) << 2) | ((kk >> 1) & 3);
        const int h2 = kk & 1;
        const int mt0 = h4 >> 4;
        _Float16* base = (_Float16*)smem;
        const int tb = ((mt0 * KT1 + kt) << 9) + (ii << 1) + h2;
#pragma unroll
        for (int j = 0; j < 4; ++j) {
            const int m = (h4 & 15) + j;
            base[tb + ((m + (g2 << 4)) << 4)] = (_Float16)v[j];
        }
    }
    __syncthreads();

    // corr_conv: 882->384 relu, 384->384 relu, 384->384 (+net+inp -> x)
    layer_gemm<KT1, 0, true >(smem, 0,      ws + OFF_W1, b1v, mt, nh, lane, LDS_T1, KTD, nullptr, nullptr, 0);
    __syncthreads();
    layer_gemm<KTD, 0, true >(smem, LDS_T1, ws + OFF_W2, b2v, mt, nh, lane, LDS_T2, KTD, nullptr, nullptr, 0);
    __syncthreads();
    layer_gemm<KTD, 1, false>(smem, LDS_T2, ws + OFF_W3, b3v, mt, nh, lane, LDS_XA, KTD, net, inp, h0);
    __syncthreads();
    // c1 branch: bsum += c1_w2 @ relu(c1_w1 @ x)
    layer_gemm<KTD, 0, true >(smem, LDS_XA, ws + OFF_W4, b4v, mt, nh, lane, LDS_U,  KTD, nullptr, nullptr, 0);
    __syncthreads();
    layer_gemm<KTD, 2, false>(smem, LDS_U,  ws + OFF_W5, b5v, mt, nh, lane, 0, 0,        nullptr, nullptr, 0);
    __syncthreads();
    // c2 branch
    layer_gemm<KTD, 0, true >(smem, LDS_XA, ws + OFF_W6, b6v, mt, nh, lane, LDS_U,  KTD, nullptr, nullptr, 0);
    __syncthreads();
    layer_gemm<KTD, 2, false>(smem, LDS_U,  ws + OFF_W7, b7v, mt, nh, lane, 0, 0,        nullptr, nullptr, 0);
    __syncthreads();

    // ---- net_out = x2 (channel-major, coalesced b128 stores) ----
    for (int it = tid; it < DIMC * (HB / 4); it += 256) {
        const int n  = it >> 4;
        const int m4 = (it & 15) << 2;
        const v4f v = *(const v4f*)((const float*)(smem + LDS_BSUM) + n * HB + m4);
        *(v4f*)(out + (size_t)n * HH + h0 + m4) = v;
    }

    // ---- heads: d = r@dW.T + dB ; w = sigmoid(r@wW.T + wB), r = relu(x2) ----
    {
        const int m     = tid & 63;
        const int which = tid >> 6;       // 0..3 -> (d,0)(d,1)(w,0)(w,1)
        const int ch    = which & 1;
        const int isw   = which >> 1;
        const float* Wp = isw ? wW : dW;
        const float* bp = isw ? wB : dB;
        const float* bs = (const float*)(smem + LDS_BSUM);
        float acc = bp[ch];
        for (int n = 0; n < DIMC; ++n)
            acc = fmaf(fmaxf(bs[n * HB + m], 0.f), Wp[ch * DIMC + n], acc);
        const float res = isw ? (1.f / (1.f + __expf(-acc))) : acc;
        out[(size_t)DIMC * HH + (size_t)isw * 2 * HH + (size_t)ch * HH + h0 + m] = res;
    }
}

// ---------------------------------------------------------------------------
extern "C" void kernel_launch(void* const* d_in, const int* in_sizes, int n_in,
                              void* d_out, int out_size, void* d_ws, size_t ws_size,
                              hipStream_t stream)
{
    (void)in_sizes; (void)n_in; (void)out_size; (void)ws_size;
    const float* net   = (const float*)d_in[0];
    const float* inp   = (const float*)d_in[1];
    const float* corr  = (const float*)d_in[2];
    // d_in[3..6] = ii, jj, kk, kw: contribute exactly 0 (scaled by kw==0 and *0.0)
    const float* cc_w1 = (const float*)d_in[7];
    const float* cc_b1 = (const float*)d_in[8];
    const float* cc_w2 = (const float*)d_in[9];
    const float* cc_b2 = (const float*)d_in[10];
    const float* cc_w3 = (const float*)d_in[11];
    const float* cc_b3 = (const float*)d_in[12];
    const float* c1_w1 = (const float*)d_in[13];
    const float* c1_b1 = (const float*)d_in[14];
    const float* c1_w2 = (const float*)d_in[15];
    const float* c1_b2 = (const float*)d_in[16];
    const float* c2_w1 = (const float*)d_in[17];
    const float* c2_b1 = (const float*)d_in[18];
    const float* c2_w2 = (const float*)d_in[19];
    const float* c2_b2 = (const float*)d_in[20];
    const float* d_wp  = (const float*)d_in[21];
    const float* d_bp  = (const float*)d_in[22];
    const float* w_wp  = (const float*)d_in[23];
    const float* w_bp  = (const float*)d_in[24];

    _Float16* ws = (_Float16*)d_ws;

    // Pre-pack weights into f16 B-fragments (idempotent, runs every call).
    pack_w_kernel<<<(NTT * KT1 * 512) / 256, 256, 0, stream>>>(cc_w1, ws + OFF_W1, CORRK, KT1);
    pack_w_kernel<<<(NTT * KTD * 512) / 256, 256, 0, stream>>>(cc_w2, ws + OFF_W2, DIMC, KTD);
    pack_w_kernel<<<(NTT * KTD * 512) / 256, 256, 0, stream>>>(cc_w3, ws + OFF_W3, DIMC, KTD);
    pack_w_kernel<<<(NTT * KTD * 512) / 256, 256, 0, stream>>>(c1_w1, ws + OFF_W4, DIMC, KTD);
    pack_w_kernel<<<(NTT * KTD * 512) / 256, 256, 0, stream>>>(c1_w2, ws + OFF_W5, DIMC, KTD);
    pack_w_kernel<<<(NTT * KTD * 512) / 256, 256, 0, stream>>>(c2_w1, ws + OFF_W6, DIMC, KTD);
    pack_w_kernel<<<(NTT * KTD * 512) / 256, 256, 0, stream>>>(c2_w2, ws + OFF_W7, DIMC, KTD);

    fused_update_kernel<<<HH / HB, 256, LDS_TOTAL, stream>>>(
        net, inp, corr, ws,
        cc_b1, cc_b2, cc_b3, c1_b1, c1_b2, c2_b1, c2_b2,
        d_wp, d_bp, w_wp, w_bp, (float*)d_out);
}
